// SO8VIT_45483703665181
// MI455X (gfx1250) — matrix operand order repaired and checked by reading, not compiled, not run on hardware
//
#include <hip/hip_runtime.h>
#include <math.h>

// ============================================================================
// SO8-gated ViT block on MI455X (gfx1250): bf16 WMMA GEMMs + fused epilogues.
// GEMM: 256x128 block tile, 64x64 wave tile (4x4 WMMA), double-buffered LDS
// filled by GLOBAL_LOAD_ASYNC_TO_LDS_B128 (ASYNCcnt-tracked DMA).
// Workspace requirement: ~760 MB (bump allocator below).
// ============================================================================

typedef unsigned short u16;
typedef unsigned int   u32;
typedef __bf16 bf16_t;
typedef bf16_t v16bf __attribute__((ext_vector_type(16)));
typedef float  v8f   __attribute__((ext_vector_type(8)));

union FragU { uint4 u[2]; v16bf v; };

static constexpr int BB = 64, NN = 197, DD = 768, NHEAD = 8, HDIM = 96, HID = 3072;
static constexpr int MROWS = BB * NN;           // 12608 token rows
static constexpr float LN_EPS = 1e-5f;

// ---------------- helpers ----------------
__device__ __forceinline__ u16 f2bf(float f){
  u32 u = __float_as_uint(f);
  u32 r = u + 0x7FFFu + ((u >> 16) & 1u);      // round-to-nearest-even
  return (u16)(r >> 16);
}
__device__ __forceinline__ float bf2f(u16 h){ return __uint_as_float(((u32)h) << 16); }

// async DMA: global -> LDS, 16B per lane, tracked by ASYNCcnt
__device__ __forceinline__ void async_b128(u32 lds_addr, const void* gptr){
  asm volatile("global_load_async_to_lds_b128 %0, %1, off"
               :: "v"(lds_addr), "v"(gptr) : "memory");
}

// two-value block reduction (sum), 256 threads = 8 wave32
__device__ __forceinline__ void blockReduce2(float& a, float& b){
  __shared__ float sa[8], sb[8];
  int lane = threadIdx.x & 31, w = threadIdx.x >> 5;
  #pragma unroll
  for (int o = 16; o; o >>= 1){ a += __shfl_xor(a, o, 32); b += __shfl_xor(b, o, 32); }
  if (lane == 0){ sa[w] = a; sb[w] = b; }
  __syncthreads();
  float ra = 0.f, rb = 0.f;
  #pragma unroll
  for (int i = 0; i < 8; ++i){ ra += sa[i]; rb += sb[i]; }
  __syncthreads();                              // safe reuse on next call
  a = ra; b = rb;
}

// ---------------- f32 -> bf16 convert ----------------
__global__ void f32_to_bf16_k(const float* __restrict__ in, u16* __restrict__ out, int n){
  int i = blockIdx.x * 256 + threadIdx.x;
  if (i < n) out[i] = f2bf(in[i]);
}

// ---------------- LayerNorm -> bf16 (one block per token row) ----------------
template<int DG>
__global__ __launch_bounds__(256) void ln_bf16_k(const float* __restrict__ x,
                                                 const float* __restrict__ g,
                                                 const float* __restrict__ b,
                                                 u16* __restrict__ out){
  constexpr int IT = DG / 256;
  size_t base = (size_t)blockIdx.x * DG;
  int tid = threadIdx.x;
  float s1 = 0.f, s2 = 0.f, xl[IT];
  #pragma unroll
  for (int it = 0; it < IT; ++it){
    float v = x[base + tid + it * 256];
    xl[it] = v; s1 += v; s2 += v * v;
  }
  blockReduce2(s1, s2);
  float m = s1 / DG, var = s2 / DG - m * m, r = rsqrtf(var + LN_EPS);
  #pragma unroll
  for (int it = 0; it < IT; ++it){
    int i = tid + it * 256;
    out[base + i] = f2bf((xl[it] - m) * r * g[i] + b[i]);
  }
}

// ---------------- bf16 WMMA GEMM: C = A(M,K) @ W(N,K)^T + bias ----------------
// 256x128 block tile, BK=32, 8 waves in a 4(M) x 2(N) grid; each wave 64x64.
// Double-buffered LDS, filled with global_load_async_to_lds_b128.
__global__ __launch_bounds__(256) void gemm_bf16_k(
    const u16* __restrict__ A, const u16* __restrict__ W,
    const float* __restrict__ bias,
    float* __restrict__ outF, u16* __restrict__ outBF,
    const float* __restrict__ resid,
    int M, int N, int K, int act)
{
  constexpr int BM = 256, BN = 128, BK = 32, STR = 40;   // pad: bank-conflict-free frags
  __shared__ u16 As[2][BM * STR];
  __shared__ u16 Bs[2][BN * STR];

  int tid  = threadIdx.x;
  int lane = tid & 31, wave = tid >> 5;
  int wm = wave & 3, wn = wave >> 2;                     // 4x2 wave grid
  int tileM = blockIdx.y * BM, tileN = blockIdx.x * BN;

  v8f acc[4][4];
  #pragma unroll
  for (int a = 0; a < 4; ++a)
    #pragma unroll
    for (int b = 0; b < 4; ++b)
      acc[a][b] = v8f{0.f,0.f,0.f,0.f,0.f,0.f,0.f,0.f};

  int kTiles = K / BK;
  int l15 = lane & 15, half = lane >> 4;

  // issue one K-tile of async DMA: A = 4 chunks/thread, B = 2 chunks/thread (16B each)
  auto issue = [&](int kt, int buf){
    int k0 = kt * BK;
    #pragma unroll
    for (int i = 0; i < 4; ++i){
      int c = tid + 256 * i;                 // 1024 chunks: row = c>>2, 16B col = c&3
      int r = c >> 2, cc = (c & 3) * 8;      // cc in u16 units
      int gm = tileM + r; if (gm >= M) gm = M - 1;   // clamp: rows >= M never stored
      u32 laddr = (u32)(uintptr_t)(&As[buf][r * STR + cc]);
      async_b128(laddr, A + (size_t)gm * K + k0 + cc);
    }
    #pragma unroll
    for (int i = 0; i < 2; ++i){
      int c = tid + 256 * i;                 // 512 chunks: row = c>>2 (0..127)
      int r = c >> 2, cc = (c & 3) * 8;      // N is a multiple of 128: always in range
      u32 laddr = (u32)(uintptr_t)(&Bs[buf][r * STR + cc]);
      async_b128(laddr, W + (size_t)(tileN + r) * K + k0 + cc);
    }
  };

  issue(0, 0);
  for (int kt = 0; kt < kTiles; ++kt){
    int cb = kt & 1;
    if (kt + 1 < kTiles){
      issue(kt + 1, cb ^ 1);                           // prefetch next tile (6 async ops)
      asm volatile("s_wait_asynccnt 0x6" ::: "memory");// current tile's DMA complete
    } else {
      asm volatile("s_wait_asynccnt 0x0" ::: "memory");
    }
    __syncthreads();                                   // all waves' DMA visible

    // ---- fragment loads per ISA 7.12.2 layouts ----
    FragU bfm[4];
    #pragma unroll
    for (int nf = 0; nf < 4; ++nf){
      const u16* rp = &Bs[cb][(wn * 64 + nf * 16 + l15) * STR];
      // B: elems 0..7 -> k = half*8 + 0..7 ; elems 8..15 -> k = 16 + half*8 + 0..7
      bfm[nf].u[0] = *(const uint4*)(rp + half * 8);
      bfm[nf].u[1] = *(const uint4*)(rp + 16 + half * 8);
    }
    #pragma unroll
    for (int mf = 0; mf < 4; ++mf){
      const u16* rp = &As[cb][(wm * 64 + mf * 16 + l15) * STR];
      // A: elems 0..7 -> k = half*8 + 0..7 ; elems 8..15 -> k = 16 + half*8 + 0..7
      FragU af;
      af.u[0] = *(const uint4*)(rp + half * 8);
      af.u[1] = *(const uint4*)(rp + 16 + half * 8);
      #pragma unroll
      for (int nf = 0; nf < 4; ++nf)
        acc[mf][nf] = __builtin_amdgcn_wmma_f32_16x16x32_bf16(
            false, af.v, false, bfm[nf].v, (short)0, acc[mf][nf], false, false);
    }
    __syncthreads();                                   // frags consumed before overwrite
  }

  // ---- epilogue: bias (+GELU) (+residual) -> f32 and/or bf16 ----
  int hi8 = (lane >> 4) * 8;
  #pragma unroll
  for (int mf = 0; mf < 4; ++mf){
    #pragma unroll
    for (int nf = 0; nf < 4; ++nf){
      int col = tileN + wn * 64 + nf * 16 + l15;
      float bv = bias ? bias[col] : 0.f;
      #pragma unroll
      for (int r = 0; r < 8; ++r){
        int m = tileM + wm * 64 + mf * 16 + hi8 + r;
        if (m < M){
          float v = acc[mf][nf][r] + bv;
          if (act == 1) v = 0.5f * v * (1.0f + erff(v * 0.70710678118f));  // exact GELU
          size_t idx = (size_t)m * N + col;
          if (resid) v += resid[idx];
          if (outF)  outF[idx]  = v;
          if (outBF) outBF[idx] = f2bf(v);
        }
      }
    }
  }
}

// ---------------- attention: one wave per (b, h, q); 8 waves / block ----------------
__global__ __launch_bounds__(256) void attn_k(const u16* __restrict__ qkv,
                                              u16* __restrict__ ctx_bf){
  __shared__ float sc[8][NN + 3];
  int wslot = threadIdx.x >> 5, lane = threadIdx.x & 31;
  int gw = blockIdx.x * 8 + wslot;                 // grid sized exactly: no stragglers
  int b = gw / (NHEAD * NN);
  int rem = gw - b * NHEAD * NN;
  int h = rem / NN;
  int q = rem - h * NN;

  size_t rowq = (size_t)(b * NN + q) * (3 * DD) + h * HDIM;
  float qv[3];
  #pragma unroll
  for (int j = 0; j < 3; ++j) qv[j] = bf2f(qkv[rowq + lane + 32 * j]);

  const float scl = 0.1020620726f;                 // 96^-0.5
  for (int k = 0; k < NN; ++k){
    size_t rowk = (size_t)(b * NN + k) * (3 * DD) + DD + h * HDIM;
    float p = 0.f;
    #pragma unroll
    for (int j = 0; j < 3; ++j) p += qv[j] * bf2f(qkv[rowk + lane + 32 * j]);
    #pragma unroll
    for (int o = 16; o; o >>= 1) p += __shfl_xor(p, o, 32);
    if (lane == 0) sc[wslot][k] = p * scl;
  }
  __syncthreads();

  float mx = -1e30f;
  for (int k = lane; k < NN; k += 32) mx = fmaxf(mx, sc[wslot][k]);
  #pragma unroll
  for (int o = 16; o; o >>= 1) mx = fmaxf(mx, __shfl_xor(mx, o, 32));
  float sum = 0.f;
  for (int k = lane; k < NN; k += 32){
    float e = __expf(sc[wslot][k] - mx);
    sc[wslot][k] = e; sum += e;
  }
  #pragma unroll
  for (int o = 16; o; o >>= 1) sum += __shfl_xor(sum, o, 32);
  float inv = 1.0f / sum;
  __syncthreads();

  #pragma unroll
  for (int j = 0; j < 3; ++j){
    int d = lane + 32 * j;
    float a = 0.f;
    for (int k = 0; k < NN; ++k){
      size_t rowv = (size_t)(b * NN + k) * (3 * DD) + 2 * DD + h * HDIM;
      a += sc[wslot][k] * bf2f(qkv[rowv + d]);
    }
    ctx_bf[(size_t)(b * NN + q) * DD + h * HDIM + d] = f2bf(a * inv);
  }
}

// ---------------- SO8 adapter (one block per token row) ----------------
// MODE 0 (D=768):  out_f = x0 + LN(y;lng,lnb) + xres   (= new residual stream)
//                  out_bf = bf16(LN2(out_f; n2g,n2b)),  e_out = err
// MODE 1 (D=3072): out_bf = bf16(LN(y) + xres),         e_out = e_in + err
template<int DG, int MODE>
__global__ __launch_bounds__(256) void adapter_k(
    const float* __restrict__ xp,   const float* __restrict__ xres,
    const float* __restrict__ x0,   const float* __restrict__ eps,
    const float* __restrict__ scale,
    const float* __restrict__ lng,  const float* __restrict__ lnb,
    const float* __restrict__ n2g,  const float* __restrict__ n2b,
    float* __restrict__ out_f, u16* __restrict__ out_bf,
    const float* __restrict__ e_in, float* __restrict__ e_out)
{
  constexpr int HG = DG / 8, IT = DG / 256;
  __shared__ float yb[DG];
  __shared__ float rotS[64];
  __shared__ float errS;
  int row = blockIdx.x, tid = threadIdx.x;
  size_t base = (size_t)row * DG;

  if (tid < 32){                                  // SO8 rotation of first 8 per group
    int g = tid >> 2, j = tid & 3;
    const float* xpg = xp + base + g * HG;
    float ev = xpg[2 * j], od = xpg[2 * j + 1];
    float ang = eps[(size_t)row * 32 + g * 4 + j] * scale[0];
    float c = cosf(ang), s = sinf(ang);
    float r0 = ev * c - od * s, r1 = ev * s + od * c;
    rotS[g * 8 + 2 * j] = r0; rotS[g * 8 + 2 * j + 1] = r1;
    float d0 = r0 - ev, d1 = r1 - od;
    float sq = d0 * d0 + d1 * d1;
    sq += __shfl_xor(sq, 1, 32); sq += __shfl_xor(sq, 2, 32);   // per-group ||.||^2
    float gn = sqrtf(sq);
    gn += __shfl_xor(gn, 4, 32); gn += __shfl_xor(gn, 8, 32); gn += __shfl_xor(gn, 16, 32);
    if (tid == 0) errS = gn * (1.0f / 32.0f);     // (1 lane/group summed) -> sum of 8 group norms / 32
  }
  __syncthreads();

  float s1 = 0.f, s2 = 0.f;
  #pragma unroll
  for (int it = 0; it < IT; ++it){
    int i = tid + it * 256;
    int g = i / HG, w = i - g * HG;
    float v = (w < 8) ? rotS[g * 8 + w] : xp[base + i];
    float y = v + xres[base + i];
    yb[i] = y; s1 += y; s2 += y * y;
  }
  blockReduce2(s1, s2);
  float mean = s1 / DG, var = s2 / DG - mean * mean;
  float rstd = rsqrtf(var + LN_EPS);

  if (MODE == 0){
    float t1 = 0.f, t2 = 0.f, xn[IT];
    #pragma unroll
    for (int it = 0; it < IT; ++it){
      int i = tid + it * 256;
      float ln = (yb[i] - mean) * rstd * lng[i] + lnb[i];
      float v = x0[base + i] + ln + xres[base + i];
      xn[it] = v; out_f[base + i] = v;
      t1 += v; t2 += v * v;
    }
    blockReduce2(t1, t2);
    float m2 = t1 / DG, v2 = t2 / DG - m2 * m2, r2 = rsqrtf(v2 + LN_EPS);
    #pragma unroll
    for (int it = 0; it < IT; ++it){
      int i = tid + it * 256;
      out_bf[base + i] = f2bf((xn[it] - m2) * r2 * n2g[i] + n2b[i]);
    }
    if (tid == 0) e_out[row] = errS;
  } else {
    #pragma unroll
    for (int it = 0; it < IT; ++it){
      int i = tid + it * 256;
      float ln = (yb[i] - mean) * rstd * lng[i] + lnb[i];
      out_bf[base + i] = f2bf(ln + xres[base + i]);
    }
    if (tid == 0) e_out[row] = e_in[row] + errS;
  }
}

// ============================================================================
extern "C" void kernel_launch(void* const* d_in, const int* in_sizes, int n_in,
                              void* d_out, int out_size, void* d_ws, size_t ws_size,
                              hipStream_t stream)
{
  (void)in_sizes; (void)n_in; (void)out_size; (void)ws_size;
  const float* x     = (const float*)d_in[0];
  const float* qkv_w = (const float*)d_in[1];
  const float* qkv_b = (const float*)d_in[2];
  const float* proj_w= (const float*)d_in[3];
  const float* proj_b= (const float*)d_in[4];
  const float* n1g   = (const float*)d_in[5];
  const float* n1b   = (const float*)d_in[6];
  const float* n2g   = (const float*)d_in[7];
  const float* n2b   = (const float*)d_in[8];
  const float* owa   = (const float*)d_in[9];
  const float* oba   = (const float*)d_in[10];
  const float* sca   = (const float*)d_in[11];
  const float* lga   = (const float*)d_in[12];
  const float* lba   = (const float*)d_in[13];
  const float* owm   = (const float*)d_in[14];
  const float* obm   = (const float*)d_in[15];
  const float* scm   = (const float*)d_in[16];
  const float* lgm   = (const float*)d_in[17];
  const float* lbm   = (const float*)d_in[18];
  const float* f1w   = (const float*)d_in[19];
  const float* f1b   = (const float*)d_in[20];
  const float* f2w   = (const float*)d_in[21];
  const float* f2b   = (const float*)d_in[22];
  const float* eps_a = (const float*)d_in[23];
  const float* eps_m = (const float*)d_in[24];

  // ---- workspace bump allocator (~760 MB total) ----
  char* ws = (char*)d_ws; size_t cur = 0;
  auto alloc = [&](size_t bytes) -> void* {
    void* p = ws + cur; cur += (bytes + 255) & ~(size_t)255; return p;
  };
  const size_t M = MROWS;
  u16*  wq_bf  = (u16*) alloc((size_t)3 * DD * DD * 2);
  u16*  wp_bf  = (u16*) alloc((size_t)DD * DD * 2);
  u16*  wa_bf  = (u16*) alloc((size_t)DD * DD * 2);
  u16*  wf1_bf = (u16*) alloc((size_t)HID * DD * 2);
  u16*  wm_bf  = (u16*) alloc((size_t)HID * HID * 2);
  u16*  wf2_bf = (u16*) alloc((size_t)DD * HID * 2);
  u16*  x1_bf  = (u16*) alloc(M * DD * 2);
  u16*  qkv_bf = (u16*) alloc(M * 3 * DD * 2);
  u16*  ctx_bf = (u16*) alloc(M * DD * 2);
  float* ctxp_f = (float*)alloc(M * DD * 4);
  u16*  ctxp_bf= (u16*) alloc(M * DD * 2);
  float* xpa_f  = (float*)alloc(M * DD * 4);
  float* xres1  = (float*)alloc(M * DD * 4);
  u16*  x2_bf  = (u16*) alloc(M * DD * 2);
  float* h_f    = (float*)alloc(M * HID * 4);
  u16*  h_bf   = (u16*) alloc(M * HID * 2);
  float* xpm_f  = (float*)alloc(M * HID * 4);
  u16*  hm_bf  = (u16*) alloc(M * HID * 2);
  float* e1     = (float*)alloc(M * 4);

  auto conv = [&](const float* in, u16* out, size_t n){
    f32_to_bf16_k<<<dim3((u32)((n + 255) / 256)), dim3(256), 0, stream>>>(in, out, (int)n);
  };
  auto gemm = [&](const u16* A, const u16* W, const float* bias,
                  float* oF, u16* oBF, const float* resid, int Ncol, int K, int act){
    dim3 g(Ncol / 128, (MROWS + 255) / 256);
    gemm_bf16_k<<<g, dim3(256), 0, stream>>>(A, W, bias, oF, oBF, resid, MROWS, Ncol, K, act);
  };

  // weights -> bf16
  conv(qkv_w, wq_bf,  (size_t)3 * DD * DD);
  conv(proj_w, wp_bf, (size_t)DD * DD);
  conv(owa,   wa_bf,  (size_t)DD * DD);
  conv(f1w,   wf1_bf, (size_t)HID * DD);
  conv(owm,   wm_bf,  (size_t)HID * HID);
  conv(f2w,   wf2_bf, (size_t)DD * HID);

  // x1 = LN1(x) -> bf16
  ln_bf16_k<DD><<<dim3(MROWS), dim3(256), 0, stream>>>(x, n1g, n1b, x1_bf);
  // qkv = x1 @ qkv_w^T + b        (bf16 out for attention)
  gemm(x1_bf, wq_bf, qkv_b, nullptr, qkv_bf, nullptr, 3 * DD, DD, 0);
  // attention -> ctx (bf16)
  attn_k<<<dim3(MROWS), dim3(256), 0, stream>>>(qkv_bf, ctx_bf);
  // ctxp = ctx @ proj_w^T + b     (f32 residual + bf16 GEMM input)
  gemm(ctx_bf, wp_bf, proj_b, ctxp_f, ctxp_bf, nullptr, DD, DD, 0);
  // xp_a = ctxp @ orth_w_a^T + b
  gemm(ctxp_bf, wa_bf, oba, xpa_f, nullptr, nullptr, DD, DD, 0);
  // adapter A: xres1 = x + LN(y) + ctxp ; x2_bf = LN2(xres1) ; e1
  adapter_k<DD, 0><<<dim3(MROWS), dim3(256), 0, stream>>>(
      xpa_f, ctxp_f, x, eps_a, sca, lga, lba, n2g, n2b, xres1, x2_bf, nullptr, e1);
  // h = gelu(x2 @ fc1^T + b)      (f32 residual + bf16 GEMM input)
  gemm(x2_bf, wf1_bf, f1b, h_f, h_bf, nullptr, HID, DD, 1);
  // xp_m = h @ orth_w_m^T + b     (the big 3072x3072 GEMM)
  gemm(h_bf, wm_bf, obm, xpm_f, nullptr, nullptr, HID, HID, 0);
  // adapter M: hm_bf = bf16(LN(y) + h) ; err_out = e1 + e2
  float* err_out = (float*)d_out + (size_t)MROWS * DD;
  adapter_k<HID, 1><<<dim3(MROWS), dim3(256), 0, stream>>>(
      xpm_f, h_f, nullptr, eps_m, scm, lgm, lbm, nullptr, nullptr,
      nullptr, hm_bf, e1, err_out);
  // final: out = xres1 + hm @ fc2^T + b
  gemm(hm_bf, wf2_bf, f2b, (float*)d_out, nullptr, xres1, DD, HID, 0);
}
